// DSRefMoENoGate_14199161881016
// MI455X (gfx1250) — compile-verified, matrix operand order checked by reading, not run on hardware
//
#include <hip/hip_runtime.h>

// ---------------- problem constants (match reference) ----------------
#define DIM     2048
#define N_EXP   16
#define INTER   1408
#define SINTER  2816   // 2 * INTER shared expert width
#define T_TOK   2048
#define TOPK    6

// ---------------- tiling ----------------
#define BM   32        // token rows per block (2 WMMA M-subtiles)
#define BN   128       // output cols per block (8 waves x 16)
#define KB   64        // K staged per barrier (2 WMMA K-substeps)
#define LDK  80        // LDS row pitch in bf16 elems (160B: 16B-aligned, bank-spread)

typedef __attribute__((ext_vector_type(16))) __bf16  v16bf;
typedef __attribute__((ext_vector_type(8)))  __bf16  v8bf;
typedef __attribute__((ext_vector_type(8)))  float   v8f;
typedef __attribute__((ext_vector_type(4)))  int     v4i;

static __device__ __forceinline__ unsigned short f2bf(float f) {
  unsigned int u = __float_as_uint(f);
  u += 0x7FFFu + ((u >> 16) & 1u);           // round-to-nearest-even
  return (unsigned short)(u >> 16);
}

// Load one WMMA bf16 operand row: two 16B chunks at K = 8*half and 16+8*half.
static __device__ __forceinline__ v16bf load_frag(const unsigned short* rowp, int half) {
  union { v16bf v; v8bf h[2]; } u;
  u.h[0] = *(const v8bf*)(rowp + 8 * half);
  u.h[1] = *(const v8bf*)(rowp + 16 + 8 * half);
  return u.v;
}

static __device__ __forceinline__ v8f zero8() {
  v8f z;
#pragma unroll
  for (int i = 0; i < 8; ++i) z[i] = 0.f;
  return z;
}

static __device__ __forceinline__ v8f bwmma(v16bf a, v16bf b, v8f c) {
  return __builtin_amdgcn_wmma_f32_16x16x32_bf16(false, a, false, b, (short)0, c, false, false);
}

// ---- gfx1250 async global->LDS staging (16B), fallback = plain b128 copy ----
#if defined(__has_builtin)
#if __has_builtin(__builtin_amdgcn_global_load_async_to_lds_b128)
#define HAVE_ASYNC_LDS 1
#endif
#endif
#ifndef HAVE_ASYNC_LDS
#define HAVE_ASYNC_LDS 0
#endif

static __device__ __forceinline__ void cp16(const unsigned short* g, unsigned short* l) {
#if HAVE_ASYNC_LDS
  __builtin_amdgcn_global_load_async_to_lds_b128(
      (__attribute__((address_space(1))) v4i*)g,
      (__attribute__((address_space(3))) v4i*)l, 0, 0);
#else
  *(uint4*)l = *(const uint4*)g;
#endif
}

static __device__ __forceinline__ void wait_stage() {
#if HAVE_ASYNC_LDS
#if defined(__has_builtin) && __has_builtin(__builtin_amdgcn_s_wait_asynccnt)
  __builtin_amdgcn_s_wait_asynccnt(0);
#else
  asm volatile("s_wait_asynccnt 0x0" ::: "memory");
#endif
#endif
}

// ---------------------------------------------------------------------------
// Kernel 0a: elementwise fp32 -> bf16 (for x)
// ---------------------------------------------------------------------------
__global__ __launch_bounds__(256) void cvt_bf16(
    const float* __restrict__ in, unsigned short* __restrict__ out, long long n4) {
  const long long i = (long long)blockIdx.x * 256 + threadIdx.x;
  if (i >= n4) return;
  const float4 f = *(const float4*)(in + i * 4);
  uint2 d;
  d.x = (unsigned int)f2bf(f.x) | ((unsigned int)f2bf(f.y) << 16);
  d.y = (unsigned int)f2bf(f.z) | ((unsigned int)f2bf(f.w) << 16);
  *(uint2*)(out + i * 4) = d;
}

// ---------------------------------------------------------------------------
// Kernel 0b: fp32 (R x C) -> bf16 transposed (C x R), per expert slice.
// Makes GEMM B-staging a pure contiguous-K b128 copy (no VALU, no transpose).
// ---------------------------------------------------------------------------
__global__ __launch_bounds__(256) void cvt_transpose(
    const float* __restrict__ in, unsigned short* __restrict__ out,
    int R, int C, long long inStride, long long outStride) {
  __shared__ unsigned short t[32][40];   // 80B pitch: bank-spread
  const int e = blockIdx.z;
  const float* ip = in + (long long)e * inStride;
  unsigned short* op = out + (long long)e * outStride;
  const int r0 = blockIdx.y * 32, c0 = blockIdx.x * 32;
  const int tid = threadIdx.x;
  {
    const int r = tid >> 3, c4 = (tid & 7) * 4;
    const float4 f = *(const float4*)(ip + (long long)(r0 + r) * C + c0 + c4);
    t[c4 + 0][r] = f2bf(f.x);
    t[c4 + 1][r] = f2bf(f.y);
    t[c4 + 2][r] = f2bf(f.z);
    t[c4 + 3][r] = f2bf(f.w);
  }
  __syncthreads();
  {
    const int row = tid >> 3, seg = (tid & 7) * 4;
    uint2 d;
    d.x = (unsigned int)t[row][seg + 0] | ((unsigned int)t[row][seg + 1] << 16);
    d.y = (unsigned int)t[row][seg + 2] | ((unsigned int)t[row][seg + 3] << 16);
    *(uint2*)(op + (long long)(c0 + row) * R + r0 + seg) = d;
  }
}

// ---------------------------------------------------------------------------
// Kernel 1: deterministic per-expert token compaction (block prefix sum).
// Lists padded to multiple of 32 (BM) with tok=0, gate=0.
// ---------------------------------------------------------------------------
__global__ __launch_bounds__(256) void moe_build_routing(
    const float* __restrict__ weights, const int* __restrict__ indices,
    int* __restrict__ tok, float* __restrict__ gate, int* __restrict__ cntPad) {
  const int e = blockIdx.x;
  const int tid = threadIdx.x;
  __shared__ int scan[256];
  __shared__ int s_base;
  if (tid == 0) s_base = 0;
  __syncthreads();

  for (int t0 = 0; t0 < T_TOK; t0 += 256) {
    const int t = t0 + tid;
    float wsum = 0.f;
    int hit = 0;
#pragma unroll
    for (int k = 0; k < TOPK; ++k) {
      if (indices[t * TOPK + k] == e) { hit = 1; wsum += weights[t * TOPK + k]; }
    }
    scan[tid] = hit;
    __syncthreads();
    for (int off = 1; off < 256; off <<= 1) {      // Hillis-Steele inclusive scan
      int v = scan[tid];
      int add = (tid >= off) ? scan[tid - off] : 0;
      __syncthreads();
      scan[tid] = v + add;
      __syncthreads();
    }
    const int incl = scan[tid];
    const int excl = incl - hit;
    const int base = s_base;
    if (hit) {
      tok[e * T_TOK + base + excl]  = t;
      gate[e * T_TOK + base + excl] = wsum;
    }
    __syncthreads();
    if (tid == 255) s_base = base + incl;
    __syncthreads();
  }
  const int cnt = s_base;
  const int pad = (cnt + (BM - 1)) & ~(BM - 1);
  for (int i = cnt + tid; i < pad; i += 256) {
    tok[e * T_TOK + i]  = 0;
    gate[e * T_TOK + i] = 0.f;
  }
  if (tid == 0) cntPad[e] = pad;
}

// ---------------------------------------------------------------------------
// Kernel 2: H = silu(Xg @ Wa) * (Xg @ Wb). All-bf16 staging, b128 copies only.
// Wa/Wb are pre-transposed N-major [N][DIM]. GATHER picks rows via tok list.
// ---------------------------------------------------------------------------
template <bool GATHER>
__global__ __launch_bounds__(256) void moe_gemm1(
    const unsigned short* __restrict__ Xbf,
    const int* __restrict__ tokAll, const int* __restrict__ cntPad,
    const unsigned short* __restrict__ WaT, const unsigned short* __restrict__ WbT,
    unsigned short* __restrict__ Hout,
    int N, long long wstride, long long hstride) {
  __shared__ __align__(16) unsigned short As [BM * LDK];
  __shared__ __align__(16) unsigned short B1s[BN * LDK];
  __shared__ __align__(16) unsigned short B3s[BN * LDK];

  const int e = blockIdx.z;
  const int mLimit = GATHER ? cntPad[e] : T_TOK;
  const int m0 = blockIdx.y * BM;
  if (m0 >= mLimit) return;                      // block-uniform
  const int nb = blockIdx.x * BN;
  const unsigned short* Wa = WaT + (long long)e * wstride;
  const unsigned short* Wb = WbT + (long long)e * wstride;
  unsigned short* Hp = Hout + (long long)e * hstride;

  const int tid = threadIdx.x, lane = tid & 31, w = tid >> 5;

  // resolve gathered A row ONCE (no per-K branching)
  const int ar = tid >> 3;              // 0..31
  const int ak = (tid & 7) * 8;         // 0,8,..,56
  int arow = m0 + ar;
  if (GATHER) arow = tokAll[e * T_TOK + m0 + ar];
  const unsigned short* Arow = Xbf + (long long)arow * DIM;

  v8f c1[2] = {zero8(), zero8()}, c3[2] = {zero8(), zero8()};

  for (int kb = 0; kb < DIM; kb += KB) {
    cp16(Arow + kb + ak, &As[ar * LDK + ak]);
#pragma unroll
    for (int it = 0; it < 4; ++it) {
      const int i = tid + it * 256;
      const int rr = i >> 3, k8 = (i & 7) * 8;
      cp16(Wa + (long long)(nb + rr) * DIM + kb + k8, &B1s[rr * LDK + k8]);
      cp16(Wb + (long long)(nb + rr) * DIM + kb + k8, &B3s[rr * LDK + k8]);
    }
    if (kb + KB < DIM)   // gfx1250 global_prefetch_b8 of next K tile
      __builtin_prefetch(Wa + (long long)(nb + (tid >> 1)) * DIM + kb + KB + (tid & 1) * 32, 0, 0);
    wait_stage();
    __syncthreads();
    {
      const int half = lane >> 4, r = lane & 15;
#pragma unroll
      for (int s = 0; s < 2; ++s) {
        const int ko = s * 32;
        const v16bf a0 = load_frag(&As[r * LDK + ko], half);
        const v16bf a1 = load_frag(&As[(16 + r) * LDK + ko], half);
        const v16bf b1 = load_frag(&B1s[(w * 16 + r) * LDK + ko], half);
        const v16bf b3 = load_frag(&B3s[(w * 16 + r) * LDK + ko], half);
        c1[0] = bwmma(a0, b1, c1[0]);
        c1[1] = bwmma(a1, b1, c1[1]);
        c3[0] = bwmma(a0, b3, c3[0]);
        c3[1] = bwmma(a1, b3, c3[1]);
      }
    }
    __syncthreads();
  }

  // fused SwiGLU epilogue -> bf16 H
  const int half = lane >> 4, nloc = lane & 15;
  const int col = nb + w * 16 + nloc;
#pragma unroll
  for (int s = 0; s < 2; ++s) {
#pragma unroll
    for (int i = 0; i < 8; ++i) {
      const int m = m0 + s * 16 + i + 8 * half;
      const float p1 = c1[s][i], p3 = c3[s][i];
      const float h = (p1 / (1.f + __expf(-p1))) * p3;
      Hp[(long long)m * N + col] = f2bf(h);
    }
  }
}

// ---------------------------------------------------------------------------
// Kernel 3: P = H @ W2T'. ROUTED: out[token] += gate*P (global_atomic_add_f32).
// Shared (!ROUTED): plain stores initialize d_out.
// ---------------------------------------------------------------------------
template <bool ROUTED>
__global__ __launch_bounds__(256) void moe_gemm2(
    const unsigned short* __restrict__ Abase,
    const int* __restrict__ tokAll, const float* __restrict__ gateAll,
    const int* __restrict__ cntPad,
    const unsigned short* __restrict__ WT, float* __restrict__ Out,
    int K, long long astride, long long wstride) {
  __shared__ __align__(16) unsigned short As[BM * LDK];
  __shared__ __align__(16) unsigned short Bs[BN * LDK];

  const int e = blockIdx.z;
  const int mLimit = ROUTED ? cntPad[e] : T_TOK;
  const int m0 = blockIdx.y * BM;
  if (m0 >= mLimit) return;
  const int nb = blockIdx.x * BN;
  const unsigned short* Ap = Abase + (long long)e * astride;  // [slot][K]
  const unsigned short* Wp = WT + (long long)e * wstride;     // [DIM][K]

  const int tid = threadIdx.x, lane = tid & 31, w = tid >> 5;
  const int ar = tid >> 3, ak = (tid & 7) * 8;
  const unsigned short* Arow = Ap + (long long)(m0 + ar) * K;

  v8f c[2] = {zero8(), zero8()};

  for (int kb = 0; kb < K; kb += KB) {
    cp16(Arow + kb + ak, &As[ar * LDK + ak]);
#pragma unroll
    for (int it = 0; it < 4; ++it) {
      const int i = tid + it * 256;
      const int rr = i >> 3, k8 = (i & 7) * 8;
      cp16(Wp + (long long)(nb + rr) * K + kb + k8, &Bs[rr * LDK + k8]);
    }
    if (kb + KB < K)
      __builtin_prefetch(Wp + (long long)(nb + (tid >> 1)) * K + kb + KB + (tid & 1) * 32, 0, 0);
    wait_stage();
    __syncthreads();
    {
      const int half = lane >> 4, r = lane & 15;
#pragma unroll
      for (int s = 0; s < 2; ++s) {
        const int ko = s * 32;
        const v16bf a0 = load_frag(&As[r * LDK + ko], half);
        const v16bf a1 = load_frag(&As[(16 + r) * LDK + ko], half);
        const v16bf b  = load_frag(&Bs[(w * 16 + r) * LDK + ko], half);
        c[0] = bwmma(a0, b, c[0]);
        c[1] = bwmma(a1, b, c[1]);
      }
    }
    __syncthreads();
  }

  const int half = lane >> 4, nloc = lane & 15;
  const int col = nb + w * 16 + nloc;
  if (ROUTED) {
    const int* tokp = tokAll + e * T_TOK;
    const float* gp = gateAll + e * T_TOK;
#pragma unroll
    for (int s = 0; s < 2; ++s)
#pragma unroll
      for (int i = 0; i < 8; ++i) {
        const int sl = m0 + s * 16 + i + 8 * half;
        const float v = gp[sl] * c[s][i];                 // gate==0 on padding
        unsafeAtomicAdd(&Out[(long long)tokp[sl] * DIM + col], v);
      }
  } else {
#pragma unroll
    for (int s = 0; s < 2; ++s)
#pragma unroll
      for (int i = 0; i < 8; ++i) {
        const int sl = m0 + s * 16 + i + 8 * half;
        Out[(long long)sl * DIM + col] = c[s][i];         // initializes d_out
      }
  }
}

// ---------------------------------------------------------------------------
extern "C" void kernel_launch(void* const* d_in, const int* in_sizes, int n_in,
                              void* d_out, int out_size, void* d_ws, size_t ws_size,
                              hipStream_t stream) {
  const float* x   = (const float*)d_in[0];
  const float* wts = (const float*)d_in[1];
  const int*   idx = (const int*)d_in[2];
  const float* W1  = (const float*)d_in[3];
  const float* W3  = (const float*)d_in[4];
  const float* W2  = (const float*)d_in[5];
  const float* Ws1 = (const float*)d_in[6];
  const float* Ws3 = (const float*)d_in[7];
  const float* Ws2 = (const float*)d_in[8];
  float* out = (float*)d_out;
  (void)in_sizes; (void)n_in; (void)out_size; (void)ws_size;

  // ---- workspace layout (bytes) ----
  char* ws = (char*)d_ws;
  size_t off = 0;
  int*   tok    = (int*)(ws + off);   off += (size_t)N_EXP * T_TOK * 4;
  float* gate   = (float*)(ws + off); off += (size_t)N_EXP * T_TOK * 4;
  int*   cntPad = (int*)(ws + off);   off += 4096;
  unsigned short* Xbf  = (unsigned short*)(ws + off); off += (size_t)T_TOK * DIM * 2;
  unsigned short* W1T  = (unsigned short*)(ws + off); off += (size_t)N_EXP * INTER * DIM * 2;
  unsigned short* W3T  = (unsigned short*)(ws + off); off += (size_t)N_EXP * INTER * DIM * 2;
  unsigned short* W2T  = (unsigned short*)(ws + off); off += (size_t)N_EXP * DIM * INTER * 2;
  unsigned short* Ws1T = (unsigned short*)(ws + off); off += (size_t)SINTER * DIM * 2;
  unsigned short* Ws3T = (unsigned short*)(ws + off); off += (size_t)SINTER * DIM * 2;
  unsigned short* Ws2T = (unsigned short*)(ws + off); off += (size_t)DIM * SINTER * 2;
  unsigned short* H    = (unsigned short*)(ws + off); off += (size_t)N_EXP * T_TOK * INTER * 2;
  unsigned short* Hs   = (unsigned short*)(ws + off); off += (size_t)T_TOK * SINTER * 2;

  const dim3 blk(256);

  // 0) one-shot precision/layout conversion: fp32 -> bf16, weights transposed N-major
  cvt_bf16<<<dim3((T_TOK * DIM) / (256 * 4)), blk, 0, stream>>>(x, Xbf, (long long)T_TOK * DIM / 4);
  cvt_transpose<<<dim3(INTER / 32, DIM / 32, N_EXP), blk, 0, stream>>>(
      W1, W1T, DIM, INTER, (long long)DIM * INTER, (long long)INTER * DIM);
  cvt_transpose<<<dim3(INTER / 32, DIM / 32, N_EXP), blk, 0, stream>>>(
      W3, W3T, DIM, INTER, (long long)DIM * INTER, (long long)INTER * DIM);
  cvt_transpose<<<dim3(DIM / 32, INTER / 32, N_EXP), blk, 0, stream>>>(
      W2, W2T, INTER, DIM, (long long)INTER * DIM, (long long)DIM * INTER);
  cvt_transpose<<<dim3(SINTER / 32, DIM / 32, 1), blk, 0, stream>>>(
      Ws1, Ws1T, DIM, SINTER, 0LL, 0LL);
  cvt_transpose<<<dim3(SINTER / 32, DIM / 32, 1), blk, 0, stream>>>(
      Ws3, Ws3T, DIM, SINTER, 0LL, 0LL);
  cvt_transpose<<<dim3(DIM / 32, SINTER / 32, 1), blk, 0, stream>>>(
      Ws2, Ws2T, SINTER, DIM, 0LL, 0LL);

  // 1) deterministic routing lists (padded to BM)
  moe_build_routing<<<dim3(N_EXP), blk, 0, stream>>>(wts, idx, tok, gate, cntPad);

  // 2) shared-expert SwiGLU: Hs = silu(x@Ws1)*(x@Ws3)
  moe_gemm1<false><<<dim3(SINTER / BN, T_TOK / BM, 1), blk, 0, stream>>>(
      Xbf, nullptr, nullptr, Ws1T, Ws3T, Hs, SINTER, 0LL, 0LL);

  // 3) routed SwiGLU on gathered tokens
  moe_gemm1<true><<<dim3(INTER / BN, T_TOK / BM, N_EXP), blk, 0, stream>>>(
      Xbf, tok, cntPad, W1T, W3T, H, INTER,
      (long long)INTER * DIM, (long long)T_TOK * INTER);

  // 4) shared down-proj initializes out: out = Hs @ Ws2
  moe_gemm2<false><<<dim3(DIM / BN, T_TOK / BM, 1), blk, 0, stream>>>(
      Hs, nullptr, nullptr, nullptr, Ws2T, out, SINTER, 0LL, 0LL);

  // 5) routed down-proj scatter-add: out[tok] += gate * (H[e] @ W2[e])
  moe_gemm2<true><<<dim3(DIM / BN, T_TOK / BM, N_EXP), blk, 0, stream>>>(
      H, tok, gate, cntPad, W2T, out, INTER,
      (long long)T_TOK * INTER, (long long)DIM * INTER);
}
